// RNNDecoder_4148938408152
// MI455X (gfx1250) — matrix operand order repaired and checked by reading, not compiled, not run on hardware
//
#include <hip/hip_runtime.h>
#include <hip/hip_bf16.h>
#include <math.h>

typedef __bf16 bf16_t;
typedef __attribute__((ext_vector_type(16))) __bf16 v16bf;
typedef __attribute__((ext_vector_type(8)))  __bf16 v8bf;
typedef __attribute__((ext_vector_type(8)))  float  v8f;

#define HID_   256
#define BB_    64
#define TT_    25
#define VOCAB_ 32000
#define KDIM_  256   // all GEMMs contract over HID_ = 256

// ---------------------------------------------------------------------------
// WMMA fragment loaders (bf16, 16x16x32 shapes; layouts per ISA 7.12.2).
// A (16x32, MxK): lanes 0-15 row M=lane, K = k0+{0..7, 16..23};
//                 lanes 16-31 row M=lane-16, K = k0+{8..15, 24..31}.
// B (32x16, KxN): stored as W[N][K] row-major (i.e. B^T). lane&15 = column,
//                 lanes 0-15 hold K=k0+{0..7, 16..23}, lanes 16-31 hold K=k0+{8..15, 24..31}.
// ---------------------------------------------------------------------------
__device__ __forceinline__ v16bf load_a_frag(const bf16_t* __restrict__ A,
                                             int lda, int m0, int k0, int lane) {
  const int row = m0 + (lane & 15);
  const int kb  = k0 + ((lane & 16) ? 8 : 0);
  const bf16_t* base = A + (size_t)row * lda + kb;
  v8bf lo = *(const v8bf*)(base);
  v8bf hi = *(const v8bf*)(base + 16);
  return __builtin_shufflevector(lo, hi, 0,1,2,3,4,5,6,7,8,9,10,11,12,13,14,15);
}

__device__ __forceinline__ v16bf load_b_frag(const bf16_t* __restrict__ W,
                                             int ldb, int n0, int k0, int lane) {
  const int col = n0 + (lane & 15);
  const int kb  = k0 + ((lane & 16) ? 8 : 0);
  const v8bf* p = (const v8bf*)(W + (size_t)col * ldb + kb);
  v8bf lo = p[0];
  v8bf hi = p[2];
  return __builtin_shufflevector(lo, hi, 0,1,2,3,4,5,6,7,8,9,10,11,12,13,14,15);
}

// ---------------------------------------------------------------------------
// Small helper kernels
// ---------------------------------------------------------------------------
extern "C" __global__ __launch_bounds__(256)
void k_cvt_bf16(const float* __restrict__ src, bf16_t* __restrict__ dst, int n) {
  int i = blockIdx.x * 256 + threadIdx.x;
  if (i < n) dst[i] = (bf16_t)src[i];
}

extern "C" __global__ __launch_bounds__(256)
void k_bias_sum(const float* __restrict__ a, const float* __restrict__ b,
                float* __restrict__ dst) {
  int i = blockIdx.x * 256 + threadIdx.x;
  if (i < 4 * HID_) dst[i] = a[i] + b[i];
}

// tokens = [BOS=1, gt[:, :-1]]; x = emb[tokens] in bf16, rows = b*T + t
extern "C" __global__ __launch_bounds__(256)
void k_embed(const int* __restrict__ gt, const float* __restrict__ emb,
             bf16_t* __restrict__ x) {
  const int rt = blockIdx.x;           // 0..1599
  const int b = rt / TT_, t = rt % TT_;
  const int tok = (t == 0) ? 1 : gt[b * TT_ + (t - 1)];
  const float* e = emb + (size_t)tok * HID_;
  const int k = threadIdx.x;           // HID_ == blockDim.x
  x[(size_t)rt * HID_ + k] = (bf16_t)e[k];
}

// h0 = image_features, c0 = 0
extern "C" __global__ __launch_bounds__(256)
void k_init_state(const float* __restrict__ imf, bf16_t* __restrict__ h0,
                  float* __restrict__ c) {
  int i = blockIdx.x * 256 + threadIdx.x;  // 64*256 = 16384
  h0[i] = (bf16_t)imf[i];
  c[i]  = 0.0f;
}

// ---------------------------------------------------------------------------
// bf16 WMMA GEMM: C[M,N] = A[M,K] * W[N,K]^T + bias[N]  (f32 out), K = 256.
// One wave computes a 32x64 tile: 2 M-frags x 4 N-frags = 8 accumulators,
// so each B fragment feeds two WMMAs. K loop fully unrolled (8 steps,
// 64 WMMAs per wave) so the compiler can pipeline loads past WMMAs.
// 8 waves / block; grid sized so tasks == blocks*8 exactly.
// M must be a multiple of 32, N a multiple of 64.
// ---------------------------------------------------------------------------
extern "C" __global__ __launch_bounds__(256)
void k_wmma_gemm(const bf16_t* __restrict__ A, const bf16_t* __restrict__ W,
                 const float* __restrict__ bias, float* __restrict__ C,
                 int N, int ngroups) {
  const int lane = threadIdx.x & 31;
  const int wave = threadIdx.x >> 5;
  const int task = blockIdx.x * 8 + wave;
  const int m0 = (task / ngroups) * 32;
  const int n0 = (task % ngroups) * 64;

  v8f acc[2][4] = {};
#pragma unroll
  for (int k0 = 0; k0 < KDIM_; k0 += 32) {
    v16bf a0 = load_a_frag(A, KDIM_, m0,      k0, lane);
    v16bf a1 = load_a_frag(A, KDIM_, m0 + 16, k0, lane);
#pragma unroll
    for (int j = 0; j < 4; ++j) {
      v16bf b = load_b_frag(W, KDIM_, n0 + j * 16, k0, lane);
      acc[0][j] = __builtin_amdgcn_wmma_f32_16x16x32_bf16(false, a0, false, b, (short)0, acc[0][j], false, false);
      acc[1][j] = __builtin_amdgcn_wmma_f32_16x16x32_bf16(false, a1, false, b, (short)0, acc[1][j], false, false);
    }
  }

#pragma unroll
  for (int i = 0; i < 2; ++i) {
    const int rbase = m0 + i * 16 + ((lane & 16) ? 8 : 0);
#pragma unroll
    for (int j = 0; j < 4; ++j) {
      const int col = n0 + j * 16 + (lane & 15);
      const float bv = bias[col];
#pragma unroll
      for (int v = 0; v < 8; ++v) {
        C[(size_t)(rbase + v) * N + col] = acc[i][j][v] + bv;
      }
    }
  }
}

// ---------------------------------------------------------------------------
// Fused LSTM step: g = h @ W_hh^T (WMMA, staged in LDS), then cell update.
// Block jb owns h-columns [jb*64, jb*64+64) across all 4 gates:
// 4 M-tiles (rows 0..63) x 16 N-tiles. 8 waves, 8 tiles per wave.
// h is ping-ponged (hin -> hout) since every block reads all h columns.
// ---------------------------------------------------------------------------
extern "C" __global__ __launch_bounds__(256)
void k_lstm_step(const bf16_t* __restrict__ hin, const bf16_t* __restrict__ Whh,
                 const float* __restrict__ xg, float* __restrict__ c,
                 bf16_t* __restrict__ hout, bf16_t* __restrict__ hs, int t) {
  __shared__ float g[BB_][4 * 64];     // 64 KB: [row][gate*64 + local h col]
  const int lane = threadIdx.x & 31;
  const int wave = threadIdx.x >> 5;
  const int jb   = blockIdx.x;         // 0..3
  const int m0   = (wave & 3) * 16;    // M tile
  const int nb   = (wave >> 2) * 8;    // first of 8 N tiles for this wave

  v8f acc[8] = {};
#pragma unroll
  for (int k0 = 0; k0 < HID_; k0 += 32) {
    v16bf a = load_a_frag(hin, HID_, m0, k0, lane);
#pragma unroll
    for (int q = 0; q < 8; ++q) {
      const int nt   = nb + q;                          // 0..15
      const int wrow = (nt >> 2) * HID_ + jb * 64 + (nt & 3) * 16;  // W_hh row
      v16bf b = load_b_frag(Whh, HID_, wrow, k0, lane);
      acc[q] = __builtin_amdgcn_wmma_f32_16x16x32_bf16(false, a, false, b, (short)0, acc[q], false, false);
    }
  }

  const int rbase = m0 + ((lane & 16) ? 8 : 0);
#pragma unroll
  for (int q = 0; q < 8; ++q) {
    const int lc = (nb + q) * 16 + (lane & 15);          // == gate*64 + local h
#pragma unroll
    for (int v = 0; v < 8; ++v) g[rbase + v][lc] = acc[q][v];
  }
  __syncthreads();

  // Elementwise cell update: 64 rows x 64 local h columns
#pragma unroll
  for (int it = 0; it < 16; ++it) {
    const int idx = it * 256 + threadIdx.x;
    const int row = idx >> 6;            // batch b
    const int hl  = idx & 63;            // local h column
    const int hcol = jb * 64 + hl;
    const float* xr = xg + ((size_t)row * TT_ + t) * (4 * HID_);
    float iv = g[row][0 * 64 + hl] + xr[0 * HID_ + hcol];
    float fv = g[row][1 * 64 + hl] + xr[1 * HID_ + hcol];
    float gv = g[row][2 * 64 + hl] + xr[2 * HID_ + hcol];
    float ov = g[row][3 * 64 + hl] + xr[3 * HID_ + hcol];
    iv = 1.0f / (1.0f + __expf(-iv));
    fv = 1.0f / (1.0f + __expf(-fv));
    gv = tanhf(gv);
    ov = 1.0f / (1.0f + __expf(-ov));
    const int ci = row * HID_ + hcol;
    const float cv = fv * c[ci] + iv * gv;
    c[ci] = cv;
    const float hv = ov * tanhf(cv);
    const bf16_t hb = (bf16_t)hv;
    hout[ci] = hb;
    hs[((size_t)row * TT_ + t) * HID_ + hcol] = hb;
  }
}

// ---------------------------------------------------------------------------
// In-place log_softmax over V=32000 per row. One block per row, wave32
// shuffle reductions + tiny LDS cross-wave combine.
// ---------------------------------------------------------------------------
extern "C" __global__ __launch_bounds__(256)
void k_logsoftmax(float* __restrict__ out) {
  float* p = out + (size_t)blockIdx.x * VOCAB_;
  const int lane = threadIdx.x & 31;
  const int wave = threadIdx.x >> 5;
  __shared__ float red[8];
  __shared__ float bc;

  float m = -3.4e38f;
  for (int i = threadIdx.x; i < VOCAB_; i += 256) m = fmaxf(m, p[i]);
#pragma unroll
  for (int off = 16; off > 0; off >>= 1) m = fmaxf(m, __shfl_xor(m, off, 32));
  if (lane == 0) red[wave] = m;
  __syncthreads();
  if (threadIdx.x == 0) {
    float mm = red[0];
    for (int w = 1; w < 8; ++w) mm = fmaxf(mm, red[w]);
    bc = mm;
  }
  __syncthreads();
  m = bc;

  float s = 0.0f;
  for (int i = threadIdx.x; i < VOCAB_; i += 256) s += __expf(p[i] - m);
#pragma unroll
  for (int off = 16; off > 0; off >>= 1) s += __shfl_xor(s, off, 32);
  if (lane == 0) red[wave] = s;
  __syncthreads();
  if (threadIdx.x == 0) {
    float ss = 0.0f;
    for (int w = 0; w < 8; ++w) ss += red[w];
    bc = logf(ss);
  }
  __syncthreads();
  const float lse = m + bc;
  for (int i = threadIdx.x; i < VOCAB_; i += 256) p[i] = p[i] - lse;
}

// ---------------------------------------------------------------------------
// Host orchestration
// ---------------------------------------------------------------------------
extern "C" void kernel_launch(void* const* d_in, const int* in_sizes, int n_in,
                              void* d_out, int out_size, void* d_ws, size_t ws_size,
                              hipStream_t stream) {
  (void)in_sizes; (void)n_in; (void)out_size; (void)ws_size;
  const float* imf   = (const float*)d_in[0];
  const int*   gt    = (const int*)  d_in[1];
  const float* emb   = (const float*)d_in[2];
  const float* W_ih  = (const float*)d_in[3];
  const float* W_hh  = (const float*)d_in[4];
  const float* b_ih  = (const float*)d_in[5];
  const float* b_hh  = (const float*)d_in[6];
  const float* W_out = (const float*)d_in[7];
  const float* b_out = (const float*)d_in[8];
  float* out = (float*)d_out;

  char* ws = (char*)d_ws;
  bf16_t* Wih_b   = (bf16_t*)(ws + 0);          // 1024*256*2 = 524288
  bf16_t* Whh_b   = (bf16_t*)(ws + 524288);     // 524288
  bf16_t* Wout_b  = (bf16_t*)(ws + 1048576);    // 32000*256*2 = 16384000
  float*  biassum = (float*) (ws + 17432576);   // 1024*4
  bf16_t* x_b     = (bf16_t*)(ws + 17436672);   // 1600*256*2 = 819200
  float*  xg      = (float*) (ws + 18255872);   // 1600*1024*4 = 6553600
  bf16_t* h0b     = (bf16_t*)(ws + 24809472);   // 64*256*2
  bf16_t* h1b     = (bf16_t*)(ws + 24842240);   // 64*256*2
  float*  cbuf    = (float*) (ws + 24875008);   // 64*256*4
  bf16_t* hs      = (bf16_t*)(ws + 24940544);   // 1600*256*2

  // Precision conversion of weights (fp32 -> bf16)
  k_cvt_bf16<<<1024, 256, 0, stream>>>(W_ih, Wih_b, 4 * HID_ * HID_);
  k_cvt_bf16<<<1024, 256, 0, stream>>>(W_hh, Whh_b, 4 * HID_ * HID_);
  k_cvt_bf16<<<32000, 256, 0, stream>>>(W_out, Wout_b, VOCAB_ * HID_);
  k_bias_sum<<<4, 256, 0, stream>>>(b_ih, b_hh, biassum);

  // Token gather + state init
  k_embed<<<BB_ * TT_, 256, 0, stream>>>(gt, emb, x_b);
  k_init_state<<<64, 256, 0, stream>>>(imf, h0b, cbuf);

  // xg = x @ W_ih^T + (b_ih + b_hh): M=1600 (50 tiles of 32), N=1024
  // -> 50*16 = 800 wave-tasks -> 100 blocks
  k_wmma_gemm<<<100, 256, 0, stream>>>(x_b, Wih_b, biassum, xg, 4 * HID_, 16);

  // Recurrent scan: 25 dependent fused step kernels, ping-pong h
  for (int t = 0; t < TT_; ++t) {
    const bf16_t* hin = (t & 1) ? h1b : h0b;
    bf16_t* hout      = (t & 1) ? h0b : h1b;
    k_lstm_step<<<4, 256, 0, stream>>>(hin, Whh_b, xg, cbuf, hout, hs, t);
  }

  // logits = hs @ W_out^T + b_out straight into d_out:
  // M=1600 (50 tiles of 32), N=32000 (500 groups of 64)
  // -> 25000 wave-tasks -> 3125 blocks
  k_wmma_gemm<<<3125, 256, 0, stream>>>(hs, Wout_b, b_out, out, VOCAB_, 500);

  // In-place log_softmax per row
  k_logsoftmax<<<BB_ * TT_, 256, 0, stream>>>(out);
}
